// ShallowLSTM_seq2seq_66236985639137
// MI455X (gfx1250) — compile-verified, matrix operand order checked by reading, not run on hardware
//
#include <hip/hip_runtime.h>
#include <hip/hip_bf16.h>
#include <math.h>

// ---------------------------------------------------------------------------
// Shapes (compile-time constants from the reference)
// ---------------------------------------------------------------------------
#define B_    256
#define TIN_  168
#define F_    128
#define H_    512
#define G4_   2048            // 4*H
#define K_    640             // F + H  (concatenated [x|h] GEMM depth)
#define TOUT_ 48

typedef __attribute__((ext_vector_type(16))) __bf16 v16bf;
typedef __attribute__((ext_vector_type(8)))  float  v8f;

__device__ __forceinline__ float sigmoidf_(float x) {
    return 1.0f / (1.0f + __expf(-x));
}

// ---------------------------------------------------------------------------
// prep: convert weights to bf16 in WMMA-friendly [N][K] layout.
//   encW/decW: [2048][640], row n = gate-row n, k<128 from Wih, k>=128 from Whh
//   wlinB:     [128][512]  (W_lin is already [F][H] = [N][K])
// ---------------------------------------------------------------------------
__global__ void prep_kernel(const float* __restrict__ eWih, const float* __restrict__ eWhh,
                            const float* __restrict__ dWih, const float* __restrict__ dWhh,
                            const float* __restrict__ Wlin,
                            __bf16* __restrict__ encW, __bf16* __restrict__ decW,
                            __bf16* __restrict__ wlinB) {
    int tid    = blockIdx.x * blockDim.x + threadIdx.x;
    int stride = gridDim.x * blockDim.x;
    const int NW = G4_ * K_;
    for (int i = tid; i < NW; i += stride) {
        int n = i / K_, k = i - n * K_;
        float ev = (k < F_) ? eWih[n * F_ + k] : eWhh[n * H_ + (k - F_)];
        float dv = (k < F_) ? dWih[n * F_ + k] : dWhh[n * H_ + (k - F_)];
        encW[i] = (__bf16)ev;
        decW[i] = (__bf16)dv;
    }
    for (int i = tid; i < F_ * H_; i += stride)
        wlinB[i] = (__bf16)Wlin[i];
}

// ---------------------------------------------------------------------------
// init: h0 = c0 = 0;  dec_in = x[:, T_IN-1, :]
// ---------------------------------------------------------------------------
__global__ void init_kernel(const float* __restrict__ x,
                            float* __restrict__ h, float* __restrict__ c,
                            float* __restrict__ dec_in) {
    int tid    = blockIdx.x * blockDim.x + threadIdx.x;
    int stride = gridDim.x * blockDim.x;
    for (int i = tid; i < B_ * H_; i += stride) { h[i] = 0.0f; c[i] = 0.0f; }
    for (int i = tid; i < B_ * F_; i += stride) {
        int b = i >> 7, f = i & (F_ - 1);
        dec_in[i] = x[(size_t)b * TIN_ * F_ + (TIN_ - 1) * F_ + f];
    }
}

// ---------------------------------------------------------------------------
// Fused LSTM step:
//   gates = [x_t | h_in](bf16) @ W^T  (W stored [2048][640] bf16)  + bias
//   c_out = sig(f)*c + sig(i)*tanh(g);  h_out = sig(o)*tanh(c_out)
//
// Grid (8,8): block (bm,bn) owns rows [32*bm,+32) and hidden slice [64*bn,+64)
// of ALL FOUR gates (256 gate columns) -> elementwise stays block-local.
// 256 threads = 8 wave32; wave w owns gate-col-tiles {2w,2w+1} x 2 row-tiles.
// h is double buffered (h_in read-only, h_out write-only) -> no inter-block
// race inside one launch.
// ---------------------------------------------------------------------------
__global__ __launch_bounds__(256)
void lstm_step_kernel(const float* __restrict__ xin, long xstride,
                      const __bf16* __restrict__ W,
                      const float* __restrict__ bias,
                      const float* __restrict__ h_in,
                      float* __restrict__ h_out,
                      float* __restrict__ c) {
    // 40KB LDS: A tile during GEMM, then reused as f32 gate buffer.
    __shared__ union __attribute__((aligned(32))) {
        __bf16 a[32][K_];     // 40960 B
        float  g[32][256];    // 32768 B
    } sm;

    const int tid  = threadIdx.x;
    const int row0 = blockIdx.x * 32;
    const int n0   = blockIdx.y * 64;

    // Stage A = [x_t | h_in] rows row0..row0+31 into LDS as bf16 [M][K]
    for (int i = tid; i < 32 * F_; i += 256) {
        int r = i >> 7, k = i & (F_ - 1);
        sm.a[r][k] = (__bf16)xin[(long)(row0 + r) * xstride + k];
    }
    for (int i = tid; i < 32 * H_; i += 256) {
        int r = i >> 9, k = i & (H_ - 1);
        sm.a[r][F_ + k] = (__bf16)h_in[(row0 + r) * H_ + k];
    }
    __syncthreads();

    const int wave = tid >> 5;          // wave32
    const int lane = tid & 31;
    const int tr   = lane & 15;         // row (A) / col (B) within 16-tile
    const int kh   = (lane >> 4) << 4;  // K half select: 0 or 16
    const int ct0  = wave * 2;          // this wave's first gate-col-tile

    v8f acc[2][2] = {};                 // [row-tile][col-tile]

    for (int kk = 0; kk < K_; kk += 32) {
        // A fragments: lane reads 16 contiguous bf16 (32B) of its row
        v16bf a0 = *(const v16bf*)&sm.a[tr][kk + kh];
        v16bf a1 = *(const v16bf*)&sm.a[16 + tr][kk + kh];
#pragma unroll
        for (int ctl = 0; ctl < 2; ++ctl) {
            int ct   = ct0 + ctl;
            int gidx = ct >> 2;                                  // which gate (i,f,g,o)
            int N    = gidx * H_ + n0 + ((ct & 3) << 4) + tr;    // global gate row
            const __bf16* bp = &W[(size_t)N * K_ + kk + kh];
            v16bf bfrag = *(const v16bf*)bp;
            if (kk + 32 < K_)
                __builtin_prefetch((const void*)(bp + 32), 0, 3); // WGP-scope prefetch
            acc[0][ctl] = __builtin_amdgcn_wmma_f32_16x16x32_bf16(
                false, a0, false, bfrag, (short)0, acc[0][ctl], false, false);
            acc[1][ctl] = __builtin_amdgcn_wmma_f32_16x16x32_bf16(
                false, a1, false, bfrag, (short)0, acc[1][ctl], false, false);
        }
    }

    __syncthreads();   // A no longer needed; reuse LDS as gate buffer

    // C layout: VGPR j, lanes 0-15 -> row j, lanes 16-31 -> row 8+j; col = lane&15
#pragma unroll
    for (int m = 0; m < 2; ++m) {
        int rbase = m * 16 + ((lane >> 4) << 3);
#pragma unroll
        for (int ctl = 0; ctl < 2; ++ctl) {
            int colb = (ct0 + ctl) * 16 + tr;   // == gate*64 + local-hidden
#pragma unroll
            for (int j = 0; j < 8; ++j)
                sm.g[rbase + j][colb] = acc[m][ctl][j];
        }
    }
    __syncthreads();

    // Elementwise LSTM cell for our (32 rows x 64 hidden) slice
    for (int e = tid; e < 32 * 64; e += 256) {
        int r  = e >> 6, hl = e & 63;
        int hg = n0 + hl;
        float iv = sm.g[r][hl]        + bias[hg];
        float fv = sm.g[r][64 + hl]   + bias[H_ + hg];
        float gv = sm.g[r][128 + hl]  + bias[2 * H_ + hg];
        float ov = sm.g[r][192 + hl]  + bias[3 * H_ + hg];
        long idx = (long)(row0 + r) * H_ + hg;
        float cn = sigmoidf_(fv) * c[idx] + sigmoidf_(iv) * tanhf(gv);
        float hn = sigmoidf_(ov) * tanhf(cn);
        c[idx]     = cn;
        h_out[idx] = hn;
    }
}

// ---------------------------------------------------------------------------
// Decoder projection: out_t = h @ W_lin^T   (256x512)@(512x128)
// Writes f32 into d_out[:, t, :] and feeds dec_in (next decoder input).
// Grid 8 (row tiles of 32), 8 waves, wave w owns col-tile w x 2 row-tiles.
// ---------------------------------------------------------------------------
__global__ __launch_bounds__(256)
void proj_kernel(const float* __restrict__ h, const __bf16* __restrict__ Wl,
                 float* __restrict__ out, float* __restrict__ dec_in, int t) {
    __shared__ __attribute__((aligned(32))) __bf16 as_[32][H_];   // 32KB

    const int tid  = threadIdx.x;
    const int row0 = blockIdx.x * 32;

    for (int i = tid; i < 32 * H_; i += 256) {
        int r = i >> 9, k = i & (H_ - 1);
        as_[r][k] = (__bf16)h[(row0 + r) * H_ + k];
    }
    __syncthreads();

    const int wave = tid >> 5;
    const int lane = tid & 31;
    const int tr   = lane & 15;
    const int kh   = (lane >> 4) << 4;

    v8f acc[2] = {};
    for (int kk = 0; kk < H_; kk += 32) {
        v16bf a0 = *(const v16bf*)&as_[tr][kk + kh];
        v16bf a1 = *(const v16bf*)&as_[16 + tr][kk + kh];
        int N = wave * 16 + tr;
        const __bf16* bp = &Wl[(size_t)N * H_ + kk + kh];
        v16bf bfrag = *(const v16bf*)bp;
        if (kk + 32 < H_)
            __builtin_prefetch((const void*)(bp + 32), 0, 3);     // WGP-scope prefetch
        acc[0] = __builtin_amdgcn_wmma_f32_16x16x32_bf16(
            false, a0, false, bfrag, (short)0, acc[0], false, false);
        acc[1] = __builtin_amdgcn_wmma_f32_16x16x32_bf16(
            false, a1, false, bfrag, (short)0, acc[1], false, false);
    }

    int col = wave * 16 + tr;
#pragma unroll
    for (int m = 0; m < 2; ++m) {
#pragma unroll
        for (int j = 0; j < 8; ++j) {
            int row = row0 + m * 16 + ((lane >> 4) << 3) + j;
            float v = acc[m][j];
            out[(size_t)row * TOUT_ * F_ + (size_t)t * F_ + col] = v;
            dec_in[row * F_ + col] = v;
        }
    }
}

// ---------------------------------------------------------------------------
// Host-side launcher
// ---------------------------------------------------------------------------
extern "C" void kernel_launch(void* const* d_in, const int* in_sizes, int n_in,
                              void* d_out, int out_size, void* d_ws, size_t ws_size,
                              hipStream_t stream) {
    const float* x       = (const float*)d_in[0];
    const float* enc_Wih = (const float*)d_in[1];
    const float* enc_Whh = (const float*)d_in[2];
    const float* enc_b   = (const float*)d_in[3];
    const float* dec_Wih = (const float*)d_in[4];
    const float* dec_Whh = (const float*)d_in[5];
    const float* dec_b   = (const float*)d_in[6];
    const float* W_lin   = (const float*)d_in[7];
    float* out = (float*)d_out;

    // Workspace layout
    char* ws = (char*)d_ws;
    __bf16* encW  = (__bf16*)ws;                       // 2048*640
    __bf16* decW  = encW + (size_t)G4_ * K_;           // 2048*640
    __bf16* wlinB = decW + (size_t)G4_ * K_;           // 128*512
    size_t bf16_bytes = ((size_t)2 * G4_ * K_ + (size_t)F_ * H_) * sizeof(__bf16);
    float* hA     = (float*)(ws + bf16_bytes);         // 256*512
    float* hB     = hA + (size_t)B_ * H_;              // 256*512
    float* cbuf   = hB + (size_t)B_ * H_;              // 256*512
    float* dec_in = cbuf + (size_t)B_ * H_;            // 256*128

    prep_kernel<<<256, 256, 0, stream>>>(enc_Wih, enc_Whh, dec_Wih, dec_Whh,
                                         W_lin, encW, decW, wlinB);
    init_kernel<<<256, 256, 0, stream>>>(x, hA, cbuf, dec_in);

    float* hcur = hA;
    float* hnxt = hB;

    // Encoder: 168 sequential fused steps
    for (int t = 0; t < TIN_; ++t) {
        lstm_step_kernel<<<dim3(8, 8), 256, 0, stream>>>(
            x + (size_t)t * F_, (long)(TIN_ * F_), encW, enc_b, hcur, hnxt, cbuf);
        float* tmp = hcur; hcur = hnxt; hnxt = tmp;
    }

    // Decoder: 48 sequential steps, output feeds back as next input
    for (int t = 0; t < TOUT_; ++t) {
        lstm_step_kernel<<<dim3(8, 8), 256, 0, stream>>>(
            dec_in, (long)F_, decW, dec_b, hcur, hnxt, cbuf);
        float* tmp = hcur; hcur = hnxt; hnxt = tmp;
        proj_kernel<<<8, 256, 0, stream>>>(hcur, wlinB, out, dec_in, t);
    }
}
